// Mutation_GAT_12232066859614
// MI455X (gfx1250) — matrix operand order repaired and checked by reading, not compiled, not run on hardware
//
#include <hip/hip_runtime.h>
#include <hip/hip_fp16.h>

// ---------------------------------------------------------------------------
// 2-layer GAT for gfx1250 (MI455X).
//  - GEMMs: v_wmma_f32_16x16x32_f16, one 16x(16*NT) strip per wave so each
//    A fragment feeds NT back-to-back WMMAs (A re-read 16x -> 4x).
//  - Edge softmax/aggregation: native f32 atomics (global_atomic_add_f32,
//    sign-partitioned int/uint atomic max for segment-max).
//  - log-softmax: one wave32 per node via __shfl_xor.
// ---------------------------------------------------------------------------

typedef _Float16 half_t;
typedef __attribute__((ext_vector_type(16))) _Float16 v16h;
typedef __attribute__((ext_vector_type(8)))  _Float16 v8h;
typedef __attribute__((ext_vector_type(8)))  float    v8f;

#define NN     50000
#define EE     800000
#define ETOT   (EE + NN)          // edges + self loops = 850000
#define FIN    256
#define HEADS_ 4
#define HID_   64
#define HD     256                // HEADS_*HID_
#define CLS    32
#define NEG    0.2f

// ---------------------------------------------------------------------------
// helpers
// ---------------------------------------------------------------------------
__device__ __forceinline__ void edge_sd(const long long* ei, int e, int& s, int& d) {
    if (e < EE) { s = (int)ei[e]; d = (int)ei[EE + e]; }
    else        { s = e - EE; d = s; }              // self loop
}

// float atomic max via sign-partitioned integer atomics (init bits = 0xFFFFFFFF)
__device__ __forceinline__ void atomicMaxF(float* addr, float v) {
    if (v >= 0.0f) atomicMax((int*)addr, __float_as_int(v));
    else           atomicMin((unsigned int*)addr, __float_as_uint(v));
}

// ---------------------------------------------------------------------------
// prep kernels
// ---------------------------------------------------------------------------
__global__ void k_cast_x(const float* __restrict__ x, half_t* __restrict__ xh) {
    int i = blockIdx.x * 256 + threadIdx.x;       // N*HD divisible by 256
    xh[i] = (half_t)x[i];
}

// W [K, Ncol] row-major (f32)  ->  Wt [Ncol, K] (f16)
__global__ void k_transpose_w(const float* __restrict__ W, half_t* __restrict__ Wt,
                              int K, int Ncol) {
    int i = blockIdx.x * 256 + threadIdx.x;
    if (i >= K * Ncol) return;
    int n = i / K, k = i - n * K;
    Wt[(size_t)n * K + k] = (half_t)W[(size_t)k * Ncol + n];
}

// ---------------------------------------------------------------------------
// WMMA GEMM: C[M,Ncol] = A[M,K=256](f16, row-major) * Bt[Ncol,K](f16)^T
// One wave computes a 16 x (16*NT) strip: A fragment loaded once per K-step,
// reused by NT consecutive WMMAs.
// ---------------------------------------------------------------------------
template <int NT, bool HALF_OUT>
__global__ void k_gemm_wmma(const half_t* __restrict__ A,
                            const half_t* __restrict__ Bt,
                            void* __restrict__ Cout,
                            int M, int K, int Ncol) {
    const int lane   = threadIdx.x & 31;
    const int wave   = threadIdx.x >> 5;
    const int groups = (Ncol >> 4) / NT;          // N-strip groups
    const int gtile  = blockIdx.x * 4 + wave;
    if (gtile >= (M >> 4) * groups) return;       // wave-uniform: EXEC stays all-1s
    const int tm = gtile / groups;
    const int tg = gtile - tm * groups;

    const int row = tm * 16 + (lane & 15);        // A row for this lane
    const int ab  = (lane >> 4) * 8;              // A K-base (ISA 16-bit A layout)
    const int bb  = (lane >> 4) * 8;              // B K-base (same layout as A)

    const half_t* Arow = A + (size_t)row * K;
    const half_t* Brow[NT];
#pragma unroll
    for (int j = 0; j < NT; ++j)
        Brow[j] = Bt + (size_t)(tg * NT * 16 + j * 16 + (lane & 15)) * K;

    v8f acc[NT];
#pragma unroll
    for (int j = 0; j < NT; ++j) acc[j] = (v8f){};

#pragma unroll
    for (int k0 = 0; k0 < 256; k0 += 32) {
        v8h alo = *(const v8h*)(Arow + k0 + ab);
        v8h ahi = *(const v8h*)(Arow + k0 + ab + 16);
        v16h a;
#pragma unroll
        for (int i = 0; i < 8; ++i) { a[i] = alo[i]; a[i + 8] = ahi[i]; }
#pragma unroll
        for (int j = 0; j < NT; ++j) {
            v8h blo = *(const v8h*)(Brow[j] + k0 + bb);
            v8h bhi = *(const v8h*)(Brow[j] + k0 + bb + 16);
            v16h b;
#pragma unroll
            for (int i = 0; i < 8; ++i) { b[i] = blo[i]; b[i + 8] = bhi[i]; }
            acc[j] = __builtin_amdgcn_wmma_f32_16x16x32_f16(
                         false, a, false, b, (short)0, acc[j], false, false);
        }
    }

    const int mrow0 = tm * 16 + (lane >> 4) * 8;
#pragma unroll
    for (int j = 0; j < NT; ++j) {
        const int col = tg * NT * 16 + j * 16 + (lane & 15);
        if (HALF_OUT) {
            half_t* C = (half_t*)Cout;
#pragma unroll
            for (int i = 0; i < 8; ++i)
                C[(size_t)(mrow0 + i) * Ncol + col] = (half_t)acc[j][i];
        } else {
            float* C = (float*)Cout;
#pragma unroll
            for (int i = 0; i < 8; ++i)
                C[(size_t)(mrow0 + i) * Ncol + col] = acc[j][i];
        }
    }
}

// ---------------------------------------------------------------------------
// layer-1 attention
// ---------------------------------------------------------------------------
__global__ void k_attdot1(const half_t* __restrict__ Hh,
                          const float* __restrict__ att_s, const float* __restrict__ att_d,
                          float* __restrict__ asrc, float* __restrict__ adst) {
    int tid = blockIdx.x * 256 + threadIdx.x;
    if (tid >= NN * HEADS_) return;
    int n = tid >> 2, h = tid & 3;
    const half_t* hp = Hh + (size_t)n * HD + h * HID_;
    float ss = 0.f, sd = 0.f;
#pragma unroll 8
    for (int c = 0; c < HID_; ++c) {
        float v = (float)hp[c];
        ss += v * att_s[h * HID_ + c];
        sd += v * att_d[h * HID_ + c];
    }
    asrc[n * 4 + h] = ss;
    adst[n * 4 + h] = sd;
}

__global__ void k_edge_max1(const long long* __restrict__ ei,
                            const float* __restrict__ asrc, const float* __restrict__ adst,
                            float* __restrict__ emax) {
    int tid = blockIdx.x * 256 + threadIdx.x;
    if (tid >= ETOT * HEADS_) return;
    int e = tid >> 2, h = tid & 3;
    int s, d; edge_sd(ei, e, s, d);
    float v = asrc[s * 4 + h] + adst[d * 4 + h];
    v = v > 0.f ? v : NEG * v;
    atomicMaxF(&emax[d * 4 + h], v);
}

__global__ void k_edge_exp1(const long long* __restrict__ ei,
                            const float* __restrict__ asrc, const float* __restrict__ adst,
                            const float* __restrict__ emax,
                            float* __restrict__ alpha, float* __restrict__ den) {
    int tid = blockIdx.x * 256 + threadIdx.x;
    if (tid >= ETOT * HEADS_) return;
    int e = tid >> 2, h = tid & 3;
    int s, d; edge_sd(ei, e, s, d);
    float v = asrc[s * 4 + h] + adst[d * 4 + h];
    v = v > 0.f ? v : NEG * v;
    float p = __expf(v - emax[d * 4 + h]);
    alpha[e * 4 + h] = p;
    atomicAdd(&den[d * 4 + h], p);
}

// one block per edge; thread t handles channel t (head = t>>6)
__global__ void k_agg1(const long long* __restrict__ ei, const half_t* __restrict__ Hh,
                       const float* __restrict__ alpha, const float* __restrict__ den,
                       float* __restrict__ out) {
    int e = blockIdx.x;
    int t = threadIdx.x;
    int h = t >> 6;
    int s, d; edge_sd(ei, e, s, d);
    float a  = alpha[e * 4 + h] / (den[d * 4 + h] + 1e-16f);
    float hv = (float)Hh[(size_t)s * HD + t];
    atomicAdd(&out[(size_t)d * HD + t], hv * a);
}

__global__ void k_relu_cast(const float* __restrict__ out1, const float* __restrict__ b1,
                            half_t* __restrict__ h2) {
    int i = blockIdx.x * 256 + threadIdx.x;
    float v = out1[i] + b1[i & 255];              // HD == 256
    h2[i] = (half_t)(v > 0.f ? v : 0.f);
}

// ---------------------------------------------------------------------------
// layer-2 attention (heads = 1, channels = 32)
// ---------------------------------------------------------------------------
__global__ void k_attdot2(const float* __restrict__ L,
                          const float* __restrict__ att_s, const float* __restrict__ att_d,
                          float* __restrict__ asrc, float* __restrict__ adst) {
    int n = blockIdx.x * 256 + threadIdx.x;
    if (n >= NN) return;
    float ss = 0.f, sd = 0.f;
#pragma unroll
    for (int c = 0; c < CLS; ++c) {
        float v = L[(size_t)n * CLS + c];
        ss += v * att_s[c];
        sd += v * att_d[c];
    }
    asrc[n] = ss;
    adst[n] = sd;
}

__global__ void k_edge_max2(const long long* __restrict__ ei,
                            const float* __restrict__ asrc, const float* __restrict__ adst,
                            float* __restrict__ emax) {
    int e = blockIdx.x * 256 + threadIdx.x;
    if (e >= ETOT) return;
    int s, d; edge_sd(ei, e, s, d);
    float v = asrc[s] + adst[d];
    v = v > 0.f ? v : NEG * v;
    atomicMaxF(&emax[d], v);
}

__global__ void k_edge_exp2(const long long* __restrict__ ei,
                            const float* __restrict__ asrc, const float* __restrict__ adst,
                            const float* __restrict__ emax,
                            float* __restrict__ alpha, float* __restrict__ den) {
    int e = blockIdx.x * 256 + threadIdx.x;
    if (e >= ETOT) return;
    int s, d; edge_sd(ei, e, s, d);
    float v = asrc[s] + adst[d];
    v = v > 0.f ? v : NEG * v;
    float p = __expf(v - emax[d]);
    alpha[e] = p;
    atomicAdd(&den[d], p);
}

// 8 edges per block; 32 lanes per edge
__global__ void k_agg2(const long long* __restrict__ ei, const float* __restrict__ L,
                       const float* __restrict__ alpha, const float* __restrict__ den,
                       float* __restrict__ out) {
    int t = threadIdx.x;
    int e = blockIdx.x * 8 + (t >> 5);
    if (e >= ETOT) return;
    int c = t & 31;
    int s, d; edge_sd(ei, e, s, d);
    float a = alpha[e] / (den[d] + 1e-16f);
    atomicAdd(&out[(size_t)d * CLS + c], L[(size_t)s * CLS + c] * a);
}

// 8 nodes per block; each 32-lane wave owns one node's 32 classes
__global__ void k_logsoftmax(const float* __restrict__ out2, const float* __restrict__ b2,
                             float* __restrict__ y) {
    int t = threadIdx.x;
    int n = blockIdx.x * 8 + (t >> 5);            // N divisible by 8
    int c = t & 31;
    float v = out2[(size_t)n * CLS + c] + b2[c];
    float m = v;
#pragma unroll
    for (int o = 16; o > 0; o >>= 1) m = fmaxf(m, __shfl_xor(m, o, 32));
    float s = __expf(v - m);
#pragma unroll
    for (int o = 16; o > 0; o >>= 1) s += __shfl_xor(s, o, 32);
    y[(size_t)n * CLS + c] = (v - m) - __logf(s);
}

// ---------------------------------------------------------------------------
// launcher
// ---------------------------------------------------------------------------
extern "C" void kernel_launch(void* const* d_in, const int* in_sizes, int n_in,
                              void* d_out, int out_size, void* d_ws, size_t ws_size,
                              hipStream_t stream) {
    (void)in_sizes; (void)n_in; (void)out_size; (void)ws_size;

    const float*     x   = (const float*)d_in[0];
    const long long* ei  = (const long long*)d_in[1];   // int64 [2, E]
    const float*     W1  = (const float*)d_in[2];
    const float*     as1 = (const float*)d_in[3];
    const float*     ad1 = (const float*)d_in[4];
    const float*     b1  = (const float*)d_in[5];
    const float*     W2  = (const float*)d_in[6];
    const float*     as2 = (const float*)d_in[7];
    const float*     ad2 = (const float*)d_in[8];
    const float*     b2  = (const float*)d_in[9];
    float* y = (float*)d_out;

    // workspace carve-up (256-B aligned slabs)
    char* w = (char*)d_ws;
    size_t off = 0;
    auto carve = [&](size_t bytes) -> char* {
        char* p = w + off;
        off = (off + bytes + 255) & ~(size_t)255;
        return p;
    };
    half_t* Xh    = (half_t*)carve((size_t)NN * HD * 2);   // f16 x; reused as relu(h) f16
    half_t* Hh    = (half_t*)carve((size_t)NN * HD * 2);   // layer-1 h (f16)
    half_t* W1t   = (half_t*)carve((size_t)HD * FIN * 2);  // [256 n][256 k]
    half_t* W2t   = (half_t*)carve((size_t)CLS * HD * 2);  // [32 n][256 k]
    float*  asrc1 = (float*)carve((size_t)NN * 4 * 4);
    float*  adst1 = (float*)carve((size_t)NN * 4 * 4);
    float*  emax1 = (float*)carve((size_t)NN * 4 * 4);
    float*  den1  = (float*)carve((size_t)NN * 4 * 4);
    float*  alpha1= (float*)carve((size_t)ETOT * 4 * 4);
    float*  OUT1  = (float*)carve((size_t)NN * HD * 4);
    float*  LGT2  = (float*)carve((size_t)NN * CLS * 4);
    float*  asrc2 = (float*)carve((size_t)NN * 4);
    float*  adst2 = (float*)carve((size_t)NN * 4);
    float*  emax2 = (float*)carve((size_t)NN * 4);
    float*  den2  = (float*)carve((size_t)NN * 4);
    float*  alpha2= (float*)carve((size_t)ETOT * 4);
    float*  OUT2  = (float*)carve((size_t)NN * CLS * 4);

    // init accumulators / sentinels (0xFFFFFFFF is the max-atomic sentinel)
    hipMemsetAsync(OUT1,  0,    (size_t)NN * HD * 4, stream);
    hipMemsetAsync(emax1, 0xFF, (size_t)NN * 4 * 4,  stream);
    hipMemsetAsync(den1,  0,    (size_t)NN * 4 * 4,  stream);
    hipMemsetAsync(OUT2,  0,    (size_t)NN * CLS * 4, stream);
    hipMemsetAsync(emax2, 0xFF, (size_t)NN * 4,      stream);
    hipMemsetAsync(den2,  0,    (size_t)NN * 4,      stream);

    // prep: f16 cast + weight transposes
    k_cast_x<<<(NN * HD) / 256, 256, 0, stream>>>(x, Xh);
    k_transpose_w<<<(FIN * HD + 255) / 256, 256, 0, stream>>>(W1, W1t, FIN, HD);
    k_transpose_w<<<(HD * CLS + 255) / 256, 256, 0, stream>>>(W2, W2t, HD, CLS);

    // layer 1: h = x @ W1 (f16 out).  16x64 strip per wave:
    // 3125 m-tiles x 4 groups = 12500 waves -> 3125 blocks of 4 waves (exact)
    k_gemm_wmma<4, true><<<3125, 128, 0, stream>>>(Xh, W1t, (void*)Hh, NN, FIN, HD);

    k_attdot1<<<(NN * HEADS_ + 255) / 256, 256, 0, stream>>>(Hh, as1, ad1, asrc1, adst1);
    k_edge_max1<<<(ETOT * HEADS_ + 255) / 256, 256, 0, stream>>>(ei, asrc1, adst1, emax1);
    k_edge_exp1<<<(ETOT * HEADS_ + 255) / 256, 256, 0, stream>>>(ei, asrc1, adst1, emax1,
                                                                 alpha1, den1);
    k_agg1<<<ETOT, 256, 0, stream>>>(ei, Hh, alpha1, den1, OUT1);
    k_relu_cast<<<(NN * HD) / 256, 256, 0, stream>>>(OUT1, b1, Xh);   // Xh reused = relu(h) f16

    // layer 2: lgt = relu(h) @ W2 (f32 out).  16x32 strip per wave:
    // 3125 m-tiles x 1 group = 3125 waves -> 782 blocks of 4 waves (guarded)
    k_gemm_wmma<2, false><<<782, 128, 0, stream>>>(Xh, W2t, (void*)LGT2, NN, HD, CLS);

    k_attdot2<<<(NN + 255) / 256, 256, 0, stream>>>(LGT2, as2, ad2, asrc2, adst2);
    k_edge_max2<<<(ETOT + 255) / 256, 256, 0, stream>>>(ei, asrc2, adst2, emax2);
    k_edge_exp2<<<(ETOT + 255) / 256, 256, 0, stream>>>(ei, asrc2, adst2, emax2, alpha2, den2);
    k_agg2<<<(ETOT + 7) / 8, 256, 0, stream>>>(ei, LGT2, alpha2, den2, OUT2);

    k_logsoftmax<<<NN / 8, 256, 0, stream>>>(OUT2, b2, y);
}
